// RadNet_52201032516249
// MI455X (gfx1250) — matrix operand order repaired and checked by reading, not compiled, not run on hardware
//
#include <hip/hip_runtime.h>

// ---------------------------------------------------------------------------
// Types for CDNA5 WMMA
// ---------------------------------------------------------------------------
typedef __bf16 bf16;
typedef __attribute__((ext_vector_type(16))) __bf16 v16bf;
typedef __attribute__((ext_vector_type(8)))  __bf16 v8bf;
typedef __attribute__((ext_vector_type(8)))  float  v8f;

#define CUTF 0.926f  // 1.852/2
#define NSAMP 512
#define NNBR  48
#define GPTS  3375   // 15^3
#define NSEG  32

// ---------------------------------------------------------------------------
// 1. Gaussian splat: ems[n,g] -> bf16 channels-last activation (ci = 1)
// ---------------------------------------------------------------------------
__global__ __launch_bounds__(128)
void splat_kernel(const float* __restrict__ pos, const float* __restrict__ Zc,
                  const float* __restrict__ nbr, const int* __restrict__ useN,
                  bf16* __restrict__ out)
{
    int n = blockIdx.x;
    __shared__ float sdx[NNBR], sdy[NNBR], sdz[NNBR], smask[NNBR];
    int t = threadIdx.x;
    if (t < NNBR) {
        float px = pos[n * 3 + 0], py = pos[n * 3 + 1], pz = pos[n * 3 + 2];
        sdx[t] = px - nbr[(n * NNBR + t) * 3 + 0];
        sdy[t] = py - nbr[(n * NNBR + t) * 3 + 1];
        sdz[t] = pz - nbr[(n * NNBR + t) * 3 + 2];
        smask[t] = (float)useN[n * NNBR + t];
    }
    __syncthreads();
    const float step = (2.0f * CUTF) / 14.0f;
    const float inv2s2 = -0.5f / (0.2f * 0.2f);
    float z = Zc[n];
    for (int g = t; g < GPTS; g += blockDim.x) {
        int i = g / 225, j = (g / 15) % 15, k = g % 15;
        float gx = -CUTF + i * step, gy = -CUTF + j * step, gz = -CUTF + k * step;
        float acc = 0.0f;
        #pragma unroll 4
        for (int m = 0; m < NNBR; ++m) {
            float dx = sdx[m] - gx, dy = sdy[m] - gy, dz = sdz[m] - gz;
            float d2 = dx * dx + dy * dy + dz * dz;
            acc += smask[m] * __expf(inv2s2 * d2);
        }
        float r = sqrtf(gx * gx + gy * gy + gz * gz);
        float filt = erfcf(r - CUTF) * 0.5f;
        out[(size_t)n * GPTS + g] = (bf16)(z * filt * acc);
    }
}

// ---------------------------------------------------------------------------
// 2. Weight packing into WMMA 16x16x32 bf16 A-fragment layout
//    dst[coTile][chunk][lane][e]; K index = tap*ci + c (taps outer)
//    A layout: lane m holds K half*16+e (contiguous)
// ---------------------------------------------------------------------------
__global__ void pack_conv_w(const float* __restrict__ w, bf16* __restrict__ dst,
                            int ci, int chunks, int total)
{
    int K = ci * 27;
    for (int idx = blockIdx.x * blockDim.x + threadIdx.x; idx < total;
         idx += gridDim.x * blockDim.x) {
        int e      = idx & 15;
        int lane   = (idx >> 4) & 31;
        int chunk  = (idx >> 9) % chunks;
        int coTile = (idx >> 9) / chunks;
        int half = lane >> 4, m = lane & 15;
        int klocal = half * 16 + e;
        int k = chunk * 32 + klocal;
        float v = 0.0f;
        if (k < K) {
            int tap = k / ci, c = k % ci;
            v = w[((size_t)(coTile * 16 + m) * ci + c) * 27 + tap];
        }
        dst[idx] = (bf16)v;
    }
}

__global__ void pack_fc1_w(const float* __restrict__ w, bf16* __restrict__ dst)
{
    // 64 oTiles x 512 chunks x 32 lanes x 16 elems; our k = s*32 + c
    // (channels-last), reference flatten k_ref = c*512 + s (NCDHW, C=32, S=512).
    int total = 64 * 512 * 512;
    for (int idx = blockIdx.x * blockDim.x + threadIdx.x; idx < total;
         idx += gridDim.x * blockDim.x) {
        int e     = idx & 15;
        int lane  = (idx >> 4) & 31;
        int chunk = (idx >> 9) & 511;
        int oTile = idx >> 18;
        int half = lane >> 4, m = lane & 15;
        int klocal = half * 16 + e;
        int k = chunk * 32 + klocal;           // < 16384 always
        int s = k >> 5, c = k & 31;
        int kref = c * 512 + s;
        dst[idx] = (bf16)w[(size_t)(oTile * 16 + m) * 16384 + kref];
    }
}

// ---------------------------------------------------------------------------
// 3. Implicit-GEMM conv3d (3x3x3, pad 1) with WMMA bf16 -> f32 accum + ELU.
//    One wave: 16 spatial outputs x (COTILES*16) output channels, one sample.
//    A-fragments for a K-window of the layer's weights are staged in LDS and
//    shared by all 8 waves of the block (ds_load_b128 in the inner loop).
//    CI templated -> all chunk->tap/channel math is shifts (no integer div).
// ---------------------------------------------------------------------------
template <int CI, int COTILES>
__global__ __launch_bounds__(256)
void conv_wmma(const bf16* __restrict__ in, bf16* __restrict__ out,
               const bf16* __restrict__ wpack, const float* __restrict__ bias,
               int Din, int Dout, int stride)
{
    constexpr int CHUNKS = (27 * CI + 31) / 32;
    constexpr int CBLK0  = (COTILES == 4) ? 16 : (COTILES == 2 ? 32 : 64);
    constexpr int CBLKW  = (CBLK0 < CHUNKS) ? CBLK0 : CHUNKS;   // K-window
    __shared__ bf16 smem[COTILES * CBLKW * 512];                // <= 64 KB

    const int co = COTILES * 16;
    int Sin3 = Din * Din * Din, Sout3 = Dout * Dout * Dout;
    int sTiles = (Sout3 + 15) >> 4;
    int wave = blockIdx.x * 8 + (threadIdx.x >> 5);   // grid is exact
    int lane = threadIdx.x & 31;
    int n = wave / sTiles;
    int sTile = wave - n * sTiles;
    int col = lane & 15, half = lane >> 4;
    int s = sTile * 16 + col;
    bool sValid = s < Sout3;
    int ss = sValid ? s : 0;
    int od = ss / (Dout * Dout);
    int rem = ss % (Dout * Dout);
    int oh = rem / Dout, ow = rem % Dout;

    v8f acc[COTILES];
    #pragma unroll
    for (int t = 0; t < COTILES; ++t) acc[t] = v8f{};

    const bf16* inBase = in + (size_t)n * Sin3 * CI;

    for (int cb = 0; cb < CHUNKS; cb += CBLKW) {
        int nb = (CHUNKS - cb) < CBLKW ? (CHUNKS - cb) : CBLKW;
        if (cb) __syncthreads();   // previous window fully consumed
        // ---- cooperative stage of A fragments for this K-window ----
        {
            int winE = nb * 512;
            int totE = COTILES * winE;
            for (int i = threadIdx.x * 16; i < totE; i += 256 * 16) {
                int t = i / winE;
                int r = i - t * winE;
                *(v16bf*)(smem + t * (CBLKW * 512) + r) =
                    *(const v16bf*)(wpack + ((size_t)t * CHUNKS + cb) * 512 + r);
            }
        }
        __syncthreads();
        // ---- compute over the window ----
        for (int cl = 0; cl < nb; ++cl) {
            int chunk = cb + cl;
            int kbase = chunk * 32 + half * 16;
            v16bf bfrag = {};
            if constexpr (CI >= 16) {
                int tap = kbase / CI;          // shift (CI constexpr pow2)
                int c0  = kbase & (CI - 1);
                bool v = sValid;
                if constexpr (CI == 16) v = v && (tap < 27);  // padded tail
                if (v) {
                    int td = tap / 9, th = (tap / 3) % 3, tw = tap % 3;
                    int id = od * stride - 1 + td;
                    int ih = oh * stride - 1 + th;
                    int iw = ow * stride - 1 + tw;
                    if ((unsigned)id < (unsigned)Din &&
                        (unsigned)ih < (unsigned)Din &&
                        (unsigned)iw < (unsigned)Din) {
                        bfrag = *(const v16bf*)(inBase +
                                  (size_t)((id * Din + ih) * Din + iw) * CI + c0);
                    }
                }
            } else {  // CI == 1 gather path (layer 0 only)
                #pragma unroll
                for (int e = 0; e < 16; ++e) {
                    int k = kbase + e;
                    float v = 0.0f;
                    if (sValid && k < 27) {
                        int td = k / 9, th = (k / 3) % 3, tw = k % 3;
                        int id = od * stride - 1 + td;
                        int ih = oh * stride - 1 + th;
                        int iw = ow * stride - 1 + tw;
                        if ((unsigned)id < (unsigned)Din &&
                            (unsigned)ih < (unsigned)Din &&
                            (unsigned)iw < (unsigned)Din)
                            v = (float)inBase[(id * Din + ih) * Din + iw];
                    }
                    bfrag[e] = (bf16)v;
                }
            }
            const bf16* sp = smem + cl * 512 + lane * 16;
            #pragma unroll
            for (int t = 0; t < COTILES; ++t) {
                v16bf afrag = *(const v16bf*)(sp + t * (CBLKW * 512));
                acc[t] = __builtin_amdgcn_wmma_f32_16x16x32_bf16(
                    false, afrag, false, bfrag, (short)0, acc[t], false, false);
            }
        }
    }

    if (!sValid) return;
    int rowBase = half * 8;
    bf16* outBase = out + (size_t)(n * Sout3 + s) * co;
    #pragma unroll
    for (int t = 0; t < COTILES; ++t) {
        v8bf o8;
        #pragma unroll
        for (int r = 0; r < 8; ++r) {
            int cc = t * 16 + rowBase + r;
            float x = acc[t][r] + bias[cc];
            x = x > 0.0f ? x : (__expf(x) - 1.0f);   // ELU
            o8[r] = (bf16)x;
        }
        *(v8bf*)(outBase + t * 16 + rowBase) = o8;
    }
}

// ---------------------------------------------------------------------------
// 4. FC1 GEMM: [512,16384] x [16384,1024]^T, WMMA bf16, f32 out + ELU
// ---------------------------------------------------------------------------
__global__ __launch_bounds__(256)
void fc1_wmma(const bf16* __restrict__ act, const bf16* __restrict__ wpack,
              const float* __restrict__ bias, float* __restrict__ outp)
{
    int wave = blockIdx.x * 8 + (threadIdx.x >> 5);   // grid exact: 2048 waves
    int oTile = wave >> 5;      // 0..63
    int nTile = wave & 31;      // 0..31
    int lane = threadIdx.x & 31;
    int col = lane & 15, half = lane >> 4;
    int nn = nTile * 16 + col;

    v8f acc = v8f{};
    const bf16* a = act + (size_t)nn * 16384;
    const bf16* wp = wpack + (size_t)oTile * 512 * 512;
    for (int chunk = 0; chunk < 512; ++chunk) {
        __builtin_prefetch((const void*)(a + chunk * 32 + 1024), 0, 0);
        v16bf bfrag = *(const v16bf*)(a + chunk * 32 + half * 16);
        v16bf afrag = *(const v16bf*)(wp + ((size_t)chunk * 32 + lane) * 16);
        acc = __builtin_amdgcn_wmma_f32_16x16x32_bf16(
            false, afrag, false, bfrag, (short)0, acc, false, false);
    }
    int rowBase = half * 8;
    float* ob = outp + (size_t)nn * 1024 + oTile * 16 + rowBase;
    #pragma unroll
    for (int r = 0; r < 8; ++r) {
        float x = acc[r] + bias[oTile * 16 + rowBase + r];
        ob[r] = x > 0.0f ? x : (__expf(x) - 1.0f);
    }
}

// ---------------------------------------------------------------------------
// 5. FC2 + segment-sum:  out[seg,6] += x[n,:] @ fc2_w[j,:] + b
// ---------------------------------------------------------------------------
__global__ void zero_out_kernel(float* o, int nElems)
{
    int i = blockIdx.x * blockDim.x + threadIdx.x;
    if (i < nElems) o[i] = 0.0f;
}

__global__ __launch_bounds__(192)
void fc2_seg_kernel(const float* __restrict__ fc1out, const float* __restrict__ w2,
                    const float* __restrict__ b2, const int* __restrict__ index,
                    float* __restrict__ out)
{
    int n = blockIdx.x;
    int j = threadIdx.x >> 5;       // 0..5
    int lane = threadIdx.x & 31;
    const float* x = fc1out + (size_t)n * 1024;
    const float* wr = w2 + j * 1024;
    float s = 0.0f;
    for (int k = lane; k < 1024; k += 32) s += x[k] * wr[k];
    #pragma unroll
    for (int off = 16; off; off >>= 1) s += __shfl_down(s, off, 32);
    if (lane == 0) atomicAdd(&out[index[n] * 6 + j], s + b2[j]);
}

// ---------------------------------------------------------------------------
// Host side
// ---------------------------------------------------------------------------
struct ConvCfg { int ci, co, Din, Dout, stride, chunks; };
static const ConvCfg g_cfg[11] = {
    { 1, 64, 15, 15, 1,  1},
    {64, 64, 15, 15, 1, 54},
    {64, 16, 15, 15, 1, 54},
    {16, 16, 15, 15, 1, 14},
    {16, 16, 15, 15, 1, 14},
    {16, 16, 15, 15, 1, 14},
    {16, 64, 15,  8, 2, 14},
    {64, 32,  8,  8, 1, 54},
    {32, 32,  8,  8, 1, 27},
    {32, 32,  8,  8, 1, 27},
    {32, 32,  8,  8, 1, 27},
};

static inline size_t alignUp256(size_t x) { return (x + 255) & ~(size_t)255; }

extern "C" void kernel_launch(void* const* d_in, const int* in_sizes, int n_in,
                              void* d_out, int out_size, void* d_ws, size_t ws_size,
                              hipStream_t stream)
{
    const float* pos   = (const float*)d_in[0];
    const float* Zc    = (const float*)d_in[1];
    const float* nbr   = (const float*)d_in[2];
    const int*   useN  = (const int*)d_in[3];
    const int*   index = (const int*)d_in[4];
    const float* convW[11];
    const float* convB[11];
    for (int i = 0; i < 11; ++i) {
        convW[i] = (const float*)d_in[5 + 2 * i];
        convB[i] = (const float*)d_in[6 + 2 * i];
    }
    const float* fc1w = (const float*)d_in[27];
    const float* fc1b = (const float*)d_in[28];
    const float* fc2w = (const float*)d_in[29];
    const float* fc2b = (const float*)d_in[30];
    float* out = (float*)d_out;

    // ---- workspace partition ----
    char* ws = (char*)d_ws;
    size_t off = 0;
    const size_t actBytes = (size_t)NSAMP * GPTS * 64 * sizeof(bf16);  // 221 MB max
    bf16* actA = (bf16*)(ws + off); off += alignUp256(actBytes);
    bf16* actB = (bf16*)(ws + off); off += alignUp256(actBytes);
    bf16* wpack[11];
    int   packTotal[11];
    for (int i = 0; i < 11; ++i) {
        int coTiles = g_cfg[i].co / 16;
        packTotal[i] = coTiles * g_cfg[i].chunks * 512;
        wpack[i] = (bf16*)(ws + off);
        off += alignUp256((size_t)packTotal[i] * sizeof(bf16));
    }
    bf16* fc1pack = (bf16*)(ws + off);
    off += alignUp256((size_t)64 * 512 * 512 * sizeof(bf16));
    float* fc1out = (float*)(ws + off);
    off += alignUp256((size_t)NSAMP * 1024 * sizeof(float));

    // ---- weight packing ----
    for (int i = 0; i < 11; ++i) {
        int blocks = (packTotal[i] + 255) / 256;
        pack_conv_w<<<blocks, 256, 0, stream>>>(convW[i], wpack[i],
                                                g_cfg[i].ci, g_cfg[i].chunks,
                                                packTotal[i]);
    }
    pack_fc1_w<<<(64 * 512 * 512) / 256, 256, 0, stream>>>(fc1w, fc1pack);

    // ---- output zero + splat ----
    zero_out_kernel<<<1, 256, 0, stream>>>(out, NSEG * 6);
    splat_kernel<<<NSAMP, 128, 0, stream>>>(pos, Zc, nbr, useN, actA);

    // ---- conv stack (ping-pong) ----
    bf16* cur = actA;
    bf16* nxt = actB;
    for (int i = 0; i < 11; ++i) {
        const ConvCfg& c = g_cfg[i];
        int Sout3 = c.Dout * c.Dout * c.Dout;
        int sTiles = (Sout3 + 15) >> 4;
        int blocks = (NSAMP * sTiles) / 8;   // exact: NSAMP % 8 == 0
        switch (i) {
        case 0:
            conv_wmma<1, 4><<<blocks, 256, 0, stream>>>(cur, nxt, wpack[i],
                convB[i], c.Din, c.Dout, c.stride);
            break;
        case 1:
            conv_wmma<64, 4><<<blocks, 256, 0, stream>>>(cur, nxt, wpack[i],
                convB[i], c.Din, c.Dout, c.stride);
            break;
        case 2:
            conv_wmma<64, 1><<<blocks, 256, 0, stream>>>(cur, nxt, wpack[i],
                convB[i], c.Din, c.Dout, c.stride);
            break;
        case 3: case 4: case 5:
            conv_wmma<16, 1><<<blocks, 256, 0, stream>>>(cur, nxt, wpack[i],
                convB[i], c.Din, c.Dout, c.stride);
            break;
        case 6:
            conv_wmma<16, 4><<<blocks, 256, 0, stream>>>(cur, nxt, wpack[i],
                convB[i], c.Din, c.Dout, c.stride);
            break;
        case 7:
            conv_wmma<64, 2><<<blocks, 256, 0, stream>>>(cur, nxt, wpack[i],
                convB[i], c.Din, c.Dout, c.stride);
            break;
        default:
            conv_wmma<32, 2><<<blocks, 256, 0, stream>>>(cur, nxt, wpack[i],
                convB[i], c.Din, c.Dout, c.stride);
            break;
        }
        bf16* t = cur; cur = nxt; nxt = t;
    }
    // after 11 layers the activations are in `cur` ([512][512 spatial][32 ch] bf16)

    // ---- FC1 (WMMA) ----
    fc1_wmma<<<256, 256, 0, stream>>>(cur, fc1pack, fc1b, fc1out);

    // ---- FC2 + segment sum ----
    fc2_seg_kernel<<<NSAMP, 192, 0, stream>>>(fc1out, fc2w, fc2b, index, out);

    (void)in_sizes; (void)n_in; (void)out_size; (void)ws_size;
}
